// RedundantAttention_6399501271463
// MI455X (gfx1250) — matrix operand order repaired and checked by reading, not compiled, not run on hardware
//
#include <hip/hip_runtime.h>

typedef _Float16 f16_t;
typedef __attribute__((ext_vector_type(16))) _Float16 v16h;
typedef __attribute__((ext_vector_type(8)))  _Float16 v8h;
typedef __attribute__((ext_vector_type(8)))  float    v8f;

#define NB 32
#define SQ 64
#define EE 512
#define LDP 520    // padded LDS stride for 512-wide panels (1040B % 256B = 16)
#define LDP1 1032  // padded LDS stride for 1024-wide panel

// ---------- WMMA fragment helpers (CDNA5 16x16x32 f16 -> f32) ----------

// A-matrix 16x32 f16 layout: lanes 0-15 rows M, contiguous K to match B:
// e=0..15 -> K=e+2*hi8 (hi8 = lane>=16 ? 8:0)
__device__ __forceinline__ int kmapA(int e, int hi8) {
  return e + 2 * hi8;
}

__device__ __forceinline__ v16h load_a(const f16_t* base, int row0, int k0, int ld) {
  int lane = threadIdx.x & 31;
  int hi8 = (lane & 16) ? 8 : 0;
  const f16_t* p = base + (row0 + (lane & 15)) * ld + k0;
  v16h a;
#pragma unroll
  for (int e = 0; e < 16; ++e) a[e] = p[kmapA(e, hi8)];
  return a;
}

// B-matrix 32x16: lane -> column n, lanes 0-15 hold K=0..15, lanes 16-31 K=16..31.
// "wt": B[k][n] = W[n][k] (torch weight layout W(out,in)), contiguous in k.
__device__ __forceinline__ v16h load_b_wt(const f16_t* Wm, int n0, int k0, int ld) {
  int lane = threadIdx.x & 31;
  int hi = (lane & 16) ? 16 : 0;
  const f16_t* p = Wm + (n0 + (lane & 15)) * ld + k0 + hi;
  v16h b;
#pragma unroll
  for (int e = 0; e < 16; ++e) b[e] = p[e];
  return b;
}

// "rm": B[k][n] = src[k*ld + n] (row-major KxN source)
__device__ __forceinline__ v16h load_b_rm(const f16_t* src, int k0, int n0, int ld) {
  int lane = threadIdx.x & 31;
  int hi = (lane & 16) ? 16 : 0;
  const f16_t* p = src + (k0 + hi) * ld + n0 + (lane & 15);
  v16h b;
#pragma unroll
  for (int e = 0; e < 16; ++e) b[e] = p[e * ld];
  return b;
}

__device__ __forceinline__ v8f wmma32(v16h a, v16h b, v8f c) {
  return __builtin_amdgcn_wmma_f32_16x16x32_f16(false, a, false, b, (short)0, c,
                                                false, false);
}

#define D_ROW(m0, i, lane) ((m0) + (i) + (((lane) & 16) ? 8 : 0))
#define D_COL(n0, lane)    ((n0) + ((lane) & 15))

__device__ __forceinline__ float wave_sum(float v) {
#pragma unroll
  for (int m = 16; m > 0; m >>= 1) v += __shfl_xor(v, m, 32);
  return v;
}
__device__ __forceinline__ float wave_max(float v) {
#pragma unroll
  for (int m = 16; m > 0; m >>= 1) v = fmaxf(v, __shfl_xor(v, m, 32));
  return v;
}

// stage a 64-row f16 panel into (padded) LDS, cooperatively, vectorized 8-wide
__device__ __forceinline__ void stage64(f16_t* dst, int ldd, const f16_t* src,
                                        int lds_, int cols) {
  const int chunks = cols >> 3;
  for (int idx = threadIdx.x; idx < 64 * chunks; idx += 256) {
    int g = idx / chunks, c = (idx - g * chunks) << 3;
    *(v8h*)(dst + g * ldd + c) = *(const v8h*)(src + g * lds_ + c);
  }
}

// ---------- conversion ----------
__global__ void k_cvt(const float* __restrict__ s, f16_t* __restrict__ d, int n) {
  for (int i = blockIdx.x * blockDim.x + threadIdx.x; i < n;
       i += gridDim.x * blockDim.x)
    d[i] = (f16_t)s[i];
}

// ---------- kernel 1: Gram -> x1_attn softmax -> x1 = attn @ x ----------
__global__ __launch_bounds__(256) void k_pairattn(const f16_t* __restrict__ xh,
                                                  float* __restrict__ attn1_out,
                                                  f16_t* __restrict__ xresh) {
  extern __shared__ char smem[];
  f16_t* sX = (f16_t*)smem;                       // 64 x LDP f16 = 66560 B
  float* sG = (float*)(smem + 66560);             // 64 x 64 f32 = 16 KB
  f16_t* sAh = (f16_t*)(smem + 66560 + 16384);    // 64 x 64 f16 = 8 KB
  const int b = blockIdx.x;
  const int w = threadIdx.x >> 5, lane = threadIdx.x & 31;
  const f16_t* xb = xh + (size_t)b * SQ * EE;

  stage64(sX, LDP, xb, EE, EE);
  __syncthreads();

  // G = x x^T  (64x64, K=512): 16 tiles, 2 per wave, operands from LDS
  for (int t = w; t < 16; t += 8) {
    int ti = t >> 2, tj = t & 3;
    v8f acc = {};
    for (int kk = 0; kk < EE; kk += 32) {
      v16h a = load_a(sX, ti * 16, kk, LDP);
      v16h bb = load_b_wt(sX, tj * 16, kk, LDP);  // B[k][g] = x[g][k]
      acc = wmma32(a, bb, acc);
    }
#pragma unroll
    for (int i = 0; i < 8; ++i)
      sG[D_ROW(ti * 16, i, lane) * SQ + D_COL(tj * 16, lane)] = acc[i];
  }
  __syncthreads();

  // softmax over g of -(n_f + n_g - 2 G[f,g]) / T ; 8 rows per wave
  const float invT = 1.0f / 13.544f;
  for (int r = 0; r < 8; ++r) {
    int f = w * 8 + r;
    float nf = sG[f * SQ + f];
    int g0 = lane, g1 = lane + 32;
    float s0 = -(nf + sG[g0 * SQ + g0] - 2.0f * sG[f * SQ + g0]) * invT;
    float s1 = -(nf + sG[g1 * SQ + g1] - 2.0f * sG[f * SQ + g1]) * invT;
    float mx = wave_max(fmaxf(s0, s1));
    float e0 = __expf(s0 - mx), e1 = __expf(s1 - mx);
    float den = wave_sum(e0 + e1);
    float p0 = e0 / den, p1 = e1 / den;
    attn1_out[((size_t)b * SQ + f) * SQ + g0] = p0;
    attn1_out[((size_t)b * SQ + f) * SQ + g1] = p1;
    sAh[f * SQ + g0] = (f16_t)p0;
    sAh[f * SQ + g1] = (f16_t)p1;
  }
  __syncthreads();

  // x1 = attn @ x  (64x512, K=64): 128 tiles, 16 per wave
  f16_t* xr = xresh + (size_t)b * SQ * 1024;
  for (int t = w; t < 128; t += 8) {
    int ti = t >> 5, tj = t & 31;
    v8f acc = {};
    for (int kk = 0; kk < SQ; kk += 32) {
      v16h a = load_a(sAh, ti * 16, kk, SQ);
      v16h bb = load_b_rm(sX, kk, tj * 16, LDP);  // B[g][e] = x[g][e]
      acc = wmma32(a, bb, acc);
    }
#pragma unroll
    for (int i = 0; i < 8; ++i)
      xr[D_ROW(ti * 16, i, lane) * 1024 + D_COL(tj * 16, lane)] = (f16_t)acc[i];
  }
}

// ---------- kernel 2: qkv = x @ Wqkv^T + bqkv ----------
__global__ __launch_bounds__(256) void k_qkv(const f16_t* __restrict__ xh,
                                             const f16_t* __restrict__ Wqkvh,
                                             const float* __restrict__ bqkv,
                                             f16_t* __restrict__ qkvh) {
  extern __shared__ char smem[];
  f16_t* sX = (f16_t*)smem;  // 64 x LDP
  const int b = blockIdx.y;
  const int w = threadIdx.x >> 5, lane = threadIdx.x & 31;
  const int n0 = blockIdx.x * 128 + w * 16;
  __builtin_prefetch(Wqkvh + (size_t)n0 * EE, 0, 1);
  stage64(sX, LDP, xh + (size_t)b * SQ * EE, EE, EE);
  __syncthreads();

  v8f z = {};
  v8f acc[4] = {z, z, z, z};
  for (int kk = 0; kk < EE; kk += 32) {
    v16h bb = load_b_wt(Wqkvh, n0, kk, EE);
#pragma unroll
    for (int ti = 0; ti < 4; ++ti) {
      v16h a = load_a(sX, ti * 16, kk, LDP);
      acc[ti] = wmma32(a, bb, acc[ti]);
    }
  }
  const int col = D_COL(n0, lane);
  const float bias = bqkv[col];
  f16_t* qb = qkvh + (size_t)b * SQ * 1536;
#pragma unroll
  for (int ti = 0; ti < 4; ++ti)
#pragma unroll
    for (int i = 0; i < 8; ++i)
      qb[D_ROW(ti * 16, i, lane) * 1536 + col] = (f16_t)(acc[ti][i] + bias);
}

// ---------- kernel 3: per (b,h) attention ----------
__global__ __launch_bounds__(256) void k_attn(const f16_t* __restrict__ qkvh,
                                              float* __restrict__ attnp,
                                              f16_t* __restrict__ oh) {
  __shared__ float sS[SQ * SQ];   // 16 KB
  __shared__ f16_t sAh[SQ * SQ];  // 8 KB
  const int h = blockIdx.x, b = blockIdx.y;
  const int w = threadIdx.x >> 5, lane = threadIdx.x & 31;
  const f16_t* qb = qkvh + (size_t)b * SQ * 1536 + h * 128;
  const f16_t* kb = qb + 512;
  const f16_t* vb = qb + 1024;

  const float scale = 0.08838834764831845f;  // 1/sqrt(128)
  for (int t = w; t < 16; t += 8) {
    int ti = t >> 2, tj = t & 3;
    v8f acc = {};
    for (int kk = 0; kk < 128; kk += 32) {
      v16h a = load_a(qb, ti * 16, kk, 1536);
      v16h bb = load_b_wt(kb, tj * 16, kk, 1536);
      acc = wmma32(a, bb, acc);
    }
#pragma unroll
    for (int i = 0; i < 8; ++i)
      sS[D_ROW(ti * 16, i, lane) * SQ + D_COL(tj * 16, lane)] = acc[i] * scale;
  }
  __syncthreads();

  float* ap = attnp + ((size_t)b * 4 + h) * SQ * SQ;
  for (int r = 0; r < 8; ++r) {
    int f = w * 8 + r;
    float s0 = sS[f * SQ + lane], s1 = sS[f * SQ + lane + 32];
    float mx = wave_max(fmaxf(s0, s1));
    float e0 = __expf(s0 - mx), e1 = __expf(s1 - mx);
    float den = wave_sum(e0 + e1);
    float p0 = e0 / den, p1 = e1 / den;
    ap[f * SQ + lane] = p0;
    ap[f * SQ + lane + 32] = p1;
    sAh[f * SQ + lane] = (f16_t)p0;
    sAh[f * SQ + lane + 32] = (f16_t)p1;
  }
  __syncthreads();

  // o = a @ v : 64x128, K=64 ; wave w owns column tile tj=w
  v8f z = {};
  v8f acc[4] = {z, z, z, z};
  for (int kk = 0; kk < SQ; kk += 32) {
    v16h bb = load_b_rm(vb, kk, w * 16, 1536);
#pragma unroll
    for (int ti = 0; ti < 4; ++ti) {
      v16h a = load_a(sAh, ti * 16, kk, SQ);
      acc[ti] = wmma32(a, bb, acc[ti]);
    }
  }
  f16_t* ob = oh + (size_t)b * SQ * EE;
  const int col = h * 128 + D_COL(w * 16, lane);
#pragma unroll
  for (int ti = 0; ti < 4; ++ti)
#pragma unroll
    for (int i = 0; i < 8; ++i)
      ob[D_ROW(ti * 16, i, lane) * EE + col] = (f16_t)acc[ti][i];
}

__global__ void k_meanheads(const float* __restrict__ attnp,
                            float* __restrict__ attn2_out) {
  for (int i = blockIdx.x * blockDim.x + threadIdx.x; i < NB * SQ * SQ;
       i += gridDim.x * blockDim.x) {
    int b = i / (SQ * SQ), r = i % (SQ * SQ);
    const float* ap = attnp + (size_t)b * 4 * SQ * SQ + r;
    attn2_out[i] = 0.25f * (ap[0] + ap[SQ * SQ] + ap[2 * SQ * SQ] + ap[3 * SQ * SQ]);
  }
}

// ---------- kernel 4: x2 = o @ Wo^T + bo -> xres[:,512:1024] ----------
__global__ __launch_bounds__(256) void k_outproj(const f16_t* __restrict__ oh,
                                                 const f16_t* __restrict__ Woh,
                                                 const float* __restrict__ bo,
                                                 f16_t* __restrict__ xresh) {
  extern __shared__ char smem[];
  f16_t* sX = (f16_t*)smem;
  const int b = blockIdx.x;
  const int w = threadIdx.x >> 5, lane = threadIdx.x & 31;
  __builtin_prefetch(Woh + (size_t)w * 16 * EE, 0, 1);
  stage64(sX, LDP, oh + (size_t)b * SQ * EE, EE, EE);
  __syncthreads();

  f16_t* xr = xresh + (size_t)b * SQ * 1024 + 512;
  for (int tjj = 0; tjj < 4; ++tjj) {
    int n0 = (w + tjj * 8) * 16;
    v8f z = {};
    v8f acc[4] = {z, z, z, z};
    for (int kk = 0; kk < EE; kk += 32) {
      v16h bb = load_b_wt(Woh, n0, kk, EE);
#pragma unroll
      for (int ti = 0; ti < 4; ++ti) {
        v16h a = load_a(sX, ti * 16, kk, LDP);
        acc[ti] = wmma32(a, bb, acc[ti]);
      }
    }
    const int col = D_COL(n0, lane);
    const float bias = bo[col];
#pragma unroll
    for (int ti = 0; ti < 4; ++ti)
#pragma unroll
      for (int i = 0; i < 8; ++i)
        xr[D_ROW(ti * 16, i, lane) * 1024 + col] = (f16_t)(acc[ti][i] + bias);
  }
}

// ---------- kernel 5: hpre = x + relu(xres @ W1^T + b1) ----------
__global__ __launch_bounds__(256) void k_fc1(const f16_t* __restrict__ xresh,
                                             const f16_t* __restrict__ W1h,
                                             const float* __restrict__ b1,
                                             const float* __restrict__ x,
                                             float* __restrict__ hpre) {
  extern __shared__ char smem[];
  f16_t* sX = (f16_t*)smem;  // 64 x LDP1
  const int b = blockIdx.x;
  const int w = threadIdx.x >> 5, lane = threadIdx.x & 31;
  __builtin_prefetch(W1h + (size_t)w * 16 * 1024, 0, 1);
  stage64(sX, LDP1, xresh + (size_t)b * SQ * 1024, 1024, 1024);
  __syncthreads();

  const float* xb = x + (size_t)b * SQ * EE;
  float* hb = hpre + (size_t)b * SQ * EE;
  for (int tjj = 0; tjj < 4; ++tjj) {
    int n0 = (w + tjj * 8) * 16;
    v8f z = {};
    v8f acc[4] = {z, z, z, z};
    for (int kk = 0; kk < 1024; kk += 32) {
      v16h bb = load_b_wt(W1h, n0, kk, 1024);
#pragma unroll
      for (int ti = 0; ti < 4; ++ti) {
        v16h a = load_a(sX, ti * 16, kk, LDP1);
        acc[ti] = wmma32(a, bb, acc[ti]);
      }
    }
    const int col = D_COL(n0, lane);
    const float bias = b1[col];
#pragma unroll
    for (int ti = 0; ti < 4; ++ti)
#pragma unroll
      for (int i = 0; i < 8; ++i) {
        int m = D_ROW(ti * 16, i, lane);
        hb[m * EE + col] = xb[m * EE + col] + fmaxf(acc[ti][i] + bias, 0.0f);
      }
  }
}

// ---------- kernel 6: hpre2 = h1 + relu(h1h @ W2^T + b2) ----------
__global__ __launch_bounds__(256) void k_fc2(const f16_t* __restrict__ h1h,
                                             const f16_t* __restrict__ W2h,
                                             const float* __restrict__ b2,
                                             const float* __restrict__ h1,
                                             float* __restrict__ hpre2) {
  extern __shared__ char smem[];
  f16_t* sX = (f16_t*)smem;
  const int b = blockIdx.x;
  const int w = threadIdx.x >> 5, lane = threadIdx.x & 31;
  __builtin_prefetch(W2h + (size_t)w * 16 * EE, 0, 1);
  stage64(sX, LDP, h1h + (size_t)b * SQ * EE, EE, EE);
  __syncthreads();

  const float* rb = h1 + (size_t)b * SQ * EE;
  float* hb = hpre2 + (size_t)b * SQ * EE;
  for (int tjj = 0; tjj < 4; ++tjj) {
    int n0 = (w + tjj * 8) * 16;
    v8f z = {};
    v8f acc[4] = {z, z, z, z};
    for (int kk = 0; kk < EE; kk += 32) {
      v16h bb = load_b_wt(W2h, n0, kk, EE);
#pragma unroll
      for (int ti = 0; ti < 4; ++ti) {
        v16h a = load_a(sX, ti * 16, kk, LDP);
        acc[ti] = wmma32(a, bb, acc[ti]);
      }
    }
    const int col = D_COL(n0, lane);
    const float bias = b2[col];
#pragma unroll
    for (int ti = 0; ti < 4; ++ti)
#pragma unroll
      for (int i = 0; i < 8; ++i) {
        int m = D_ROW(ti * 16, i, lane);
        hb[m * EE + col] = rb[m * EE + col] + fmaxf(acc[ti][i] + bias, 0.0f);
      }
  }
}

// ---------- layernorm: one wave per 512-wide row ----------
__global__ __launch_bounds__(256) void k_ln(const float* __restrict__ src,
                                            float* __restrict__ dstf,
                                            f16_t* __restrict__ dsth,
                                            const float* __restrict__ g,
                                            const float* __restrict__ be) {
  const int row = blockIdx.x * 8 + (threadIdx.x >> 5);
  const int lane = threadIdx.x & 31;
  if (row >= NB * SQ) return;
  const float* r = src + (size_t)row * EE;
  float s = 0.0f, s2 = 0.0f;
  for (int j = lane; j < EE; j += 32) {
    float v = r[j];
    s += v;
    s2 += v * v;
  }
  s = wave_sum(s);
  s2 = wave_sum(s2);
  const float mean = s * (1.0f / EE);
  const float var = s2 * (1.0f / EE) - mean * mean;
  const float inv = rsqrtf(var + 1e-5f);
  for (int j = lane; j < EE; j += 32) {
    float v = (r[j] - mean) * inv * g[j] + be[j];
    dstf[(size_t)row * EE + j] = v;
    if (dsth) dsth[(size_t)row * EE + j] = (f16_t)v;
  }
}

// ---------- pair MLP: relu(relu((x_g-x_f)*w @ dW1^T + db1) . dW2 + db2) ----------
__global__ __launch_bounds__(256) void k_pair(const f16_t* __restrict__ xh,
                                              const float* __restrict__ attn,
                                              const f16_t* __restrict__ dW1h,
                                              const float* __restrict__ db1,
                                              const float* __restrict__ dW2,
                                              const float* __restrict__ db2,
                                              float* __restrict__ outd) {
  extern __shared__ char smem[];
  f16_t* sA = (f16_t*)smem;                      // 64 x LDP f16 = 66560 B
  float* sH = (float*)(smem + 66560);            // 64 x 128 f32 = 32768 B
  float* sW = (float*)(smem + 66560 + 32768);    // 64 f32
  const int f = blockIdx.x, b = blockIdx.y;
  const int w = threadIdx.x >> 5, lane = threadIdx.x & 31;
  const f16_t* xb = xh + (size_t)b * SQ * EE;
  const f16_t* xf = xb + f * EE;
  __builtin_prefetch(dW1h + (size_t)w * 16 * EE, 0, 1);
  if (threadIdx.x < SQ)
    sW[threadIdx.x] = attn[((size_t)b * SQ + f) * SQ + threadIdx.x];
  __syncthreads();

  // build A[g,e] = (x[g,e]-x[f,e])*w[g] ONCE into LDS (packed f16 math)
  for (int idx = threadIdx.x; idx < 64 * (EE / 8); idx += 256) {
    int g = idx / (EE / 8), c = (idx - g * (EE / 8)) << 3;
    v8h xg = *(const v8h*)(xb + g * EE + c);
    v8h xfv = *(const v8h*)(xf + c);
    f16_t wg = (f16_t)sW[g];
    *(v8h*)(sA + g * LDP + c) = (xg - xfv) * wg;
  }
  __syncthreads();

  // h = relu(A @ dW1^T + db1) : 64x128, K=512 ; wave w owns column tile n0=w*16
  const int n0 = w * 16;
  v8f z = {};
  v8f acc[4] = {z, z, z, z};
  for (int kk = 0; kk < EE; kk += 32) {
    v16h bb = load_b_wt(dW1h, n0, kk, EE);
#pragma unroll
    for (int ti = 0; ti < 4; ++ti) {
      v16h a = load_a(sA, ti * 16, kk, LDP);
      acc[ti] = wmma32(a, bb, acc[ti]);
    }
  }
  const int col = D_COL(n0, lane);
  const float bias = db1[col];
#pragma unroll
  for (int ti = 0; ti < 4; ++ti)
#pragma unroll
    for (int i = 0; i < 8; ++i)
      sH[D_ROW(ti * 16, i, lane) * 128 + col] = fmaxf(acc[ti][i] + bias, 0.0f);
  __syncthreads();

  // out[g] = relu(h[g,:].dW2 + db2)
  const float b2s = db2[0];
  for (int r = 0; r < 8; ++r) {
    int gg = w * 8 + r;
    float s = 0.0f;
    for (int j = lane; j < 128; j += 32) s += sH[gg * 128 + j] * dW2[j];
    s = wave_sum(s);
    if (lane == 0)
      outd[((size_t)b * SQ + f) * SQ + gg] = fmaxf(s + b2s, 0.0f);
  }
}

// ---------- host ----------
extern "C" void kernel_launch(void* const* d_in, const int* in_sizes, int n_in,
                              void* d_out, int out_size, void* d_ws, size_t ws_size,
                              hipStream_t stream) {
  (void)in_sizes; (void)n_in; (void)out_size; (void)ws_size;
  const float* x    = (const float*)d_in[0];
  const float* Wqkv = (const float*)d_in[1];
  const float* bqkv = (const float*)d_in[2];
  const float* Wo   = (const float*)d_in[3];
  const float* bo   = (const float*)d_in[4];
  const float* W1   = (const float*)d_in[5];
  const float* b1   = (const float*)d_in[6];
  const float* W2   = (const float*)d_in[7];
  const float* b2   = (const float*)d_in[8];
  const float* g1   = (const float*)d_in[9];
  const float* be1  = (const float*)d_in[10];
  const float* g2   = (const float*)d_in[11];
  const float* be2  = (const float*)d_in[12];
  const float* d1W1 = (const float*)d_in[13];
  const float* d1b1 = (const float*)d_in[14];
  const float* d1W2 = (const float*)d_in[15];
  const float* d1b2 = (const float*)d_in[16];
  const float* d2W1 = (const float*)d_in[17];
  const float* d2b1 = (const float*)d_in[18];
  const float* d2W2 = (const float*)d_in[19];
  const float* d2b2 = (const float*)d_in[20];

  float* outp  = (float*)d_out;
  float* o_out = outp;                                   // (N,S,E)
  float* o_a1  = outp + (size_t)NB * SQ * EE;            // x1_attn
  float* o_a2  = o_a1 + (size_t)NB * SQ * SQ;            // x2_attn
  float* o_d1  = o_a2 + (size_t)NB * SQ * SQ;            // diff1
  float* o_d2  = o_d1 + (size_t)NB * SQ * SQ;            // diff2

  char* wp = (char*)d_ws;
  auto alloc = [&](size_t bytes) {
    char* r = wp;
    wp += (bytes + 255) & ~(size_t)255;
    return r;
  };
  const size_t NSE = (size_t)NB * SQ * EE;
  f16_t* xh     = (f16_t*)alloc(NSE * 2);
  f16_t* Wqkvh  = (f16_t*)alloc((size_t)1536 * EE * 2);
  f16_t* Woh    = (f16_t*)alloc((size_t)EE * EE * 2);
  f16_t* W1h    = (f16_t*)alloc((size_t)EE * 1024 * 2);
  f16_t* W2h    = (f16_t*)alloc((size_t)EE * EE * 2);
  f16_t* dW1h0  = (f16_t*)alloc((size_t)128 * EE * 2);
  f16_t* dW1h1  = (f16_t*)alloc((size_t)128 * EE * 2);
  f16_t* qkvh   = (f16_t*)alloc((size_t)NB * SQ * 1536 * 2);
  f16_t* oh     = (f16_t*)alloc(NSE * 2);
  f16_t* xresh  = (f16_t*)alloc((size_t)NB * SQ * 1024 * 2);
  float* hpre1  = (float*)alloc(NSE * 4);   // becomes h1 after in-place LN
  f16_t* h1h    = (f16_t*)alloc(NSE * 2);
  float* hpre2  = (float*)alloc(NSE * 4);
  float* attnp  = (float*)alloc((size_t)NB * 4 * SQ * SQ * 4);

  const int T = 256;
  // dynamic LDS sizes
  const int SM_PANEL = 64 * LDP * 2;                       // 66560
  const int SM_PAIRATTN = SM_PANEL + 64 * 64 * 4 + 64 * 64 * 2;  // 91136
  const int SM_FC1 = 64 * LDP1 * 2;                        // 132096
  const int SM_PAIR = SM_PANEL + 64 * 128 * 4 + 64 * 4;    // 99584

  // 0) f32 -> f16 conversions
  k_cvt<<<512, T, 0, stream>>>(x, xh, (int)NSE);
  k_cvt<<<512, T, 0, stream>>>(Wqkv, Wqkvh, 1536 * EE);
  k_cvt<<<512, T, 0, stream>>>(Wo, Woh, EE * EE);
  k_cvt<<<512, T, 0, stream>>>(W1, W1h, EE * 1024);
  k_cvt<<<512, T, 0, stream>>>(W2, W2h, EE * EE);
  k_cvt<<<128, T, 0, stream>>>(d1W1, dW1h0, 128 * EE);
  k_cvt<<<128, T, 0, stream>>>(d2W1, dW1h1, 128 * EE);

  // 1) pair-distance attention + x1
  k_pairattn<<<NB, T, SM_PAIRATTN, stream>>>(xh, o_a1, xresh);
  // 2) qkv projection
  k_qkv<<<dim3(12, NB), T, SM_PANEL, stream>>>(xh, Wqkvh, bqkv, qkvh);
  // 3) attention per (b,h) + head mean
  k_attn<<<dim3(4, NB), T, 0, stream>>>(qkvh, attnp, oh);
  k_meanheads<<<256, T, 0, stream>>>(attnp, o_a2);
  // 4) output projection -> xres[:,512:]
  k_outproj<<<NB, T, SM_PANEL, stream>>>(oh, Woh, bo, xresh);
  // 5) fc1 + residual, LN1
  k_fc1<<<NB, T, SM_FC1, stream>>>(xresh, W1h, b1, x, hpre1);
  k_ln<<<NB * SQ / 8, T, 0, stream>>>(hpre1, hpre1, h1h, g1, be1);
  // 6) fc2 + residual, LN2 -> out
  k_fc2<<<NB, T, SM_PANEL, stream>>>(h1h, W2h, b2, hpre1, hpre2);
  k_ln<<<NB * SQ / 8, T, 0, stream>>>(hpre2, o_out, (f16_t*)nullptr, g2, be2);
  // 7) pair MLPs (dominant compute)
  k_pair<<<dim3(SQ, NB), T, SM_PAIR, stream>>>(xh, o_a1, dW1h0, d1b1, d1W2, d1b2, o_d1);
  k_pair<<<dim3(SQ, NB), T, SM_PAIR, stream>>>(xh, o_a2, dW1h1, d2b1, d2W2, d2b2, o_d2);
}
